// UnitaryExpert_82497731821704
// MI455X (gfx1250) — matrix operand order repaired and checked by reading, not compiled, not run on hardware
//
#include <hip/hip_runtime.h>

// ---------------------------------------------------------------------------
// UnitaryExpert fused kernel for gfx1250 (MI455X), wave32 + WMMA bf16.
//   U1 = Cayley-Neumann(W1), U2 = Cayley-Neumann(W2)      (bf16 WMMA, f32 acc)
//   Y  = rot(X @ U1^T, theta) @ U2^T + passthrough        (fused, h in LDS)
// ---------------------------------------------------------------------------

typedef __bf16 v16bf __attribute__((ext_vector_type(16)));
typedef float  v8f   __attribute__((ext_vector_type(8)));

#define DM   1024            // d_model == d_ff
#define EPSP 1e-6f           // gradient passthrough epsilon

__device__ __forceinline__ unsigned short f2bf(float x) {
  unsigned int u = __float_as_uint(x);
  u += 0x7fffu + ((u >> 16) & 1u);          // round-to-nearest-even
  return (unsigned short)(u >> 16);
}

union BFrag {                 // 16 bf16 = 8 VGPRs (A or B operand, wave32)
  v16bf          v;
  unsigned short h[16];
  unsigned int   w[8];
  uint4          q[2];
};
union CFrag {                 // 16x16 f32 C/D tile slice: 8 VGPRs per lane
  v8f   v;
  float f[8];
};

// ---------------------------------------------------------------------------
// Passthrough scalar: eps * (mean(W1) + mean(W2) + theta)
// ---------------------------------------------------------------------------
__global__ __launch_bounds__(256) void reduce_partial_kernel(
    const float* __restrict__ W1, const float* __restrict__ W2,
    float* __restrict__ partial) {
  __shared__ float sm[256];
  float s = 0.f;
  const unsigned total = 2u * DM * DM;
  for (unsigned idx = blockIdx.x * 256u + threadIdx.x; idx < total; idx += 256u * 256u)
    s += (idx < (unsigned)(DM * DM)) ? W1[idx] : W2[idx - DM * DM];
  sm[threadIdx.x] = s;
  __syncthreads();
  for (int off = 128; off > 0; off >>= 1) {
    if ((int)threadIdx.x < off) sm[threadIdx.x] += sm[threadIdx.x + off];
    __syncthreads();
  }
  if (threadIdx.x == 0) partial[blockIdx.x] = sm[0];
}

__global__ __launch_bounds__(256) void reduce_final_kernel(
    const float* __restrict__ partial, const float* __restrict__ theta,
    float* __restrict__ outscalar) {
  __shared__ float sm[256];
  sm[threadIdx.x] = partial[threadIdx.x];
  __syncthreads();
  for (int off = 128; off > 0; off >>= 1) {
    if ((int)threadIdx.x < off) sm[threadIdx.x] += sm[threadIdx.x + off];
    __syncthreads();
  }
  if (threadIdx.x == 0)
    outscalar[0] = EPSP * (sm[0] / (float)(DM * DM) + theta[0]);
}

// ---------------------------------------------------------------------------
// A = 0.5*(W - W^T);  S = I + A (f32);  Abf = Pbf = bf16(A)
// ---------------------------------------------------------------------------
__global__ __launch_bounds__(256) void skew_init_kernel(
    const float* __restrict__ W, unsigned short* __restrict__ Abf,
    unsigned short* __restrict__ Pbf, float* __restrict__ S) {
  const int idx = blockIdx.x * 256 + threadIdx.x;
  const int i = idx >> 10, j = idx & 1023;
  const float a = 0.5f * (W[(size_t)i * DM + j] - W[(size_t)j * DM + i]);
  const unsigned short ab = f2bf(a);
  Abf[idx] = ab;
  Pbf[idx] = ab;
  S[idx] = a + (i == j ? 1.0f : 0.0f);
}

// S += Pn;  Pbf = bf16(Pn)
__global__ __launch_bounds__(256) void accum_kernel(
    float* __restrict__ S, const float* __restrict__ Pn,
    unsigned short* __restrict__ Pbf) {
  const int idx = blockIdx.x * 256 + threadIdx.x;
  const float p = Pn[idx];
  S[idx] += p;
  Pbf[idx] = f2bf(p);
}

__global__ __launch_bounds__(256) void cast_bf_kernel(
    const float* __restrict__ X, unsigned short* __restrict__ Xbf) {
  const int idx = blockIdx.x * 256 + threadIdx.x;
  Xbf[idx] = f2bf(X[idx]);
}

// ---------------------------------------------------------------------------
// C = P @ A          (ADD_IDENT==0)   B[k][n] = A[k][n] = -A[n][k]   (skew!)
// C = P @ (I + A)    (ADD_IDENT==1)   diagonal of I+A is exactly 1.0
// Branchless, register-only fragment fixup (no addressable union -> no LDS
// demotion of the operand fragments).
// One 16x16 output tile per wave; 8 waves per block; 512 blocks = 4096 tiles.
// ---------------------------------------------------------------------------
template <int ADD_IDENT>
__global__ __launch_bounds__(256) void neumann_gemm_kernel(
    const unsigned short* __restrict__ Pbf,   // left operand, row-major [M][K] bf16
    const unsigned short* __restrict__ Skew,  // skew A, row-major bf16
    float* __restrict__ C) {
  const int lane = threadIdx.x & 31;
  const int wave = threadIdx.x >> 5;
  const int tile = blockIdx.x * 8 + wave;
  const int row0 = (tile >> 6) << 4;
  const int col0 = (tile & 63) << 4;
  const int r = lane & 15, hi = lane >> 4;

  CFrag acc;
#pragma unroll
  for (int i = 0; i < 8; ++i) acc.f[i] = 0.f;

  for (int kb = 0; kb < DM; kb += 32) {
    // A frag: lane half 'hi' holds K in {hi*16..hi*16+15} (contiguous)
    BFrag a;
    const uint4* pa = reinterpret_cast<const uint4*>(
        Pbf + (size_t)(row0 + r) * DM + kb + hi * 16);
    a.q[0] = pa[0];
    a.q[1] = pa[1];                       // +8 shorts
    // B frag: lanes 0-15 K=kb..kb+15, lanes 16-31 K=kb+16..kb+31 (contiguous)
    BFrag b;
    const uint4* pb = reinterpret_cast<const uint4*>(
        Skew + (size_t)(col0 + r) * DM + kb + hi * 16);
    b.q[0] = pb[0];
    b.q[1] = pb[1];
    // fragment element e (K = kb + hi*16 + e) hits the diagonal when e == d:
    const int d = (col0 + r) - (kb + hi * 16);
#pragma unroll
    for (int t = 0; t < 8; ++t) {
      unsigned wv = b.w[t] ^ 0x80008000u;            // A[k][n] = -A[n][k]
      if (ADD_IDENT) {                               // compile-time branch
        // runtime compares lower to v_cmp + v_cndmask, fragment stays in VGPRs
        wv = (d == 2 * t)     ? ((wv & 0xFFFF0000u) | 0x00003F80u) : wv;
        wv = (d == 2 * t + 1) ? ((wv & 0x0000FFFFu) | 0x3F800000u) : wv;
      }
      b.w[t] = wv;
    }
    acc.v = __builtin_amdgcn_wmma_f32_16x16x32_bf16(
        false, a.v, false, b.v, (short)0, acc.v, false, false);
  }
#pragma unroll
  for (int i = 0; i < 8; ++i)
    C[(size_t)(row0 + i + hi * 8) * DM + (col0 + r)] = acc.f[i];
}

// ---------------------------------------------------------------------------
// Fused main kernel: 16 rows of X per block (8 waves).
//   LDS tile <- bf16(X rows)        (32 KB, padded)
//   GEMM1: wave w -> h cols [128w,128w+128), 8 v8f accumulators
//   in-register pair rotation via shfl_xor (feature pair == lane pair)
//   LDS tile <- bf16(h)             (reuse, barrier-protected)
//   GEMM2 -> Y + passthrough scalar
// ---------------------------------------------------------------------------
__global__ __launch_bounds__(256) void unitary_main_kernel(
    const float* __restrict__ X, const unsigned short* __restrict__ U1,
    const unsigned short* __restrict__ U2, const float* __restrict__ theta_p,
    const float* __restrict__ pass_p, float* __restrict__ Y) {
  __shared__ unsigned short tile[16][1032];   // stride pad: 1032*2 B, 16B-aligned
  const int t = threadIdx.x;
  const int lane = t & 31, wave = t >> 5;
  const int r = lane & 15, hi = lane >> 4;
  const int m0 = blockIdx.x << 4;
  const int nbase = wave * 128;

  // ---- stage X tile -> bf16 in LDS (each thread: 64 floats of one row) ----
  {
    const int rr = t >> 4;
    const int c0 = (t & 15) * 64;
    const float4* src =
        reinterpret_cast<const float4*>(X + (size_t)(m0 + rr) * DM + c0);
    uint4* dst = reinterpret_cast<uint4*>(&tile[rr][c0]);
#pragma unroll
    for (int q = 0; q < 16; q += 2) {
      const float4 f0 = src[q], f1 = src[q + 1];
      uint4 p;
      p.x = (unsigned)f2bf(f0.x) | ((unsigned)f2bf(f0.y) << 16);
      p.y = (unsigned)f2bf(f0.z) | ((unsigned)f2bf(f0.w) << 16);
      p.z = (unsigned)f2bf(f1.x) | ((unsigned)f2bf(f1.y) << 16);
      p.w = (unsigned)f2bf(f1.z) | ((unsigned)f2bf(f1.w) << 16);
      dst[q >> 1] = p;
    }
  }
  __syncthreads();

  CFrag acc[8];
#pragma unroll
  for (int s = 0; s < 8; ++s)
#pragma unroll
    for (int i = 0; i < 8; ++i) acc[s].f[i] = 0.f;

  // ---- GEMM1: h = X @ U1^T  (B[k][n] = U1[n][k], contiguous row loads) ----
  for (int kb = 0; kb < DM; kb += 32) {
    BFrag a;
    const uint4* pa = reinterpret_cast<const uint4*>(&tile[r][kb + hi * 16]);
    a.q[0] = pa[0];
    a.q[1] = pa[1];
#pragma unroll
    for (int s = 0; s < 8; ++s) {
      BFrag b;
      const uint4* pb = reinterpret_cast<const uint4*>(
          U1 + (size_t)(nbase + s * 16 + r) * DM + kb + hi * 16);
      b.q[0] = pb[0];
      b.q[1] = pb[1];
      acc[s].v = __builtin_amdgcn_wmma_f32_16x16x32_bf16(
          false, a.v, false, b.v, (short)0, acc[s].v, false, false);
    }
  }

  // ---- quantum rotation on adjacent feature pairs (== adjacent lanes) ----
  const float th = theta_p[0];
  const float cth = cosf(th), sth = sinf(th);
#pragma unroll
  for (int s = 0; s < 8; ++s)
#pragma unroll
    for (int i = 0; i < 8; ++i) {
      const float v0 = acc[s].f[i];
      const float v1 = __shfl_xor(v0, 1, 32);       // partner feature
      acc[s].f[i] = (lane & 1) ? (sth * v1 + cth * v0)   // r1 =  s*x0 + c*x1
                               : (cth * v0 - sth * v1);  // r0 =  c*x0 - s*x1
    }

  __syncthreads();   // all waves done reading X from LDS
#pragma unroll
  for (int s = 0; s < 8; ++s)
#pragma unroll
    for (int i = 0; i < 8; ++i)
      tile[i + hi * 8][nbase + s * 16 + r] = f2bf(acc[s].f[i]);
  __syncthreads();

  // ---- GEMM2: Y = h @ U2^T ----
#pragma unroll
  for (int s = 0; s < 8; ++s)
#pragma unroll
    for (int i = 0; i < 8; ++i) acc[s].f[i] = 0.f;

  for (int kb = 0; kb < DM; kb += 32) {
    BFrag a;
    const uint4* pa = reinterpret_cast<const uint4*>(&tile[r][kb + hi * 16]);
    a.q[0] = pa[0];
    a.q[1] = pa[1];
#pragma unroll
    for (int s = 0; s < 8; ++s) {
      BFrag b;
      const uint4* pb = reinterpret_cast<const uint4*>(
          U2 + (size_t)(nbase + s * 16 + r) * DM + kb + hi * 16);
      b.q[0] = pb[0];
      b.q[1] = pb[1];
      acc[s].v = __builtin_amdgcn_wmma_f32_16x16x32_bf16(
          false, a.v, false, b.v, (short)0, acc[s].v, false, false);
    }
  }

  const float pass = pass_p[0];
#pragma unroll
  for (int s = 0; s < 8; ++s)
#pragma unroll
    for (int i = 0; i < 8; ++i)
      Y[(size_t)(m0 + i + hi * 8) * DM + (nbase + s * 16 + r)] =
          acc[s].f[i] + pass;
}

// ---------------------------------------------------------------------------
extern "C" void kernel_launch(void* const* d_in, const int* in_sizes, int n_in,
                              void* d_out, int out_size, void* d_ws,
                              size_t ws_size, hipStream_t stream) {
  (void)in_sizes; (void)n_in; (void)out_size; (void)ws_size;
  const float* X     = (const float*)d_in[0];   // [65536,1024]
  const float* W1    = (const float*)d_in[1];   // [1024,1024]
  const float* W2    = (const float*)d_in[2];   // [1024,1024]
  const float* theta = (const float*)d_in[3];   // scalar
  float*       Y     = (float*)d_out;           // [65536,1024]

  // workspace carve-out (~16 MiB)
  char* ws = (char*)d_ws;
  size_t off = 0;
  auto carve = [&](size_t bytes) -> char* {
    char* p = ws + off;
    off += (bytes + 255) & ~(size_t)255;
    return p;
  };
  const size_t MM = (size_t)DM * DM;
  unsigned short* Abf  = (unsigned short*)carve(MM * 2);
  unsigned short* Pbf  = (unsigned short*)carve(MM * 2);
  float*          Pn   = (float*)carve(MM * 4);
  float*          S    = (float*)carve(MM * 4);
  unsigned short* U1bf = (unsigned short*)carve(MM * 2);
  unsigned short* U2bf = (unsigned short*)carve(MM * 2);
  float*          part = (float*)carve(256 * 4);
  float*          scal = (float*)carve(4);

  // passthrough scalar
  reduce_partial_kernel<<<256, 256, 0, stream>>>(W1, W2, part);
  reduce_final_kernel<<<1, 256, 0, stream>>>(part, theta, scal);

  // Cayley-Neumann for both weight matrices (terms = 6)
  const float* Wm[2] = {W1, W2};
  unsigned short* Um[2] = {U1bf, U2bf};
  for (int m = 0; m < 2; ++m) {
    skew_init_kernel<<<4096, 256, 0, stream>>>(Wm[m], Abf, Pbf, S);
    for (int it = 0; it < 4; ++it) {               // adds A^2..A^5 into S
      neumann_gemm_kernel<0><<<512, 256, 0, stream>>>(Pbf, Abf, Pn);
      accum_kernel<<<4096, 256, 0, stream>>>(S, Pn, Pbf);
    }
    cast_bf_kernel<<<4096, 256, 0, stream>>>(S, Pbf);              // Pbf = bf16(S)
    neumann_gemm_kernel<1><<<512, 256, 0, stream>>>(Pbf, Abf, Pn); // U = S(I+A)
    cast_bf_kernel<<<4096, 256, 0, stream>>>(Pn, Um[m]);
  }

  // fused main pass: 65536 rows / 16 per block
  unitary_main_kernel<<<65536 / 16, 256, 0, stream>>>(X, U1bf, U2bf, theta,
                                                      scal, Y);
}
